// _AltAttention_28707561406474
// MI455X (gfx1250) — matrix operand order repaired and checked by reading, not compiled, not run on hardware
//
#include <hip/hip_runtime.h>
#include <hip/hip_bf16.h>
#include <stdint.h>

typedef _Float16 v16h __attribute__((ext_vector_type(16)));
typedef _Float16 v8h  __attribute__((ext_vector_type(8)));
typedef float    v8f  __attribute__((ext_vector_type(8)));

constexpr int Bb = 4;
constexpr int Nn = 2048;
constexpr int Cc = 768;
constexpr int Hh = 12;
constexpr int Dd = 64;
constexpr int C3 = 3 * Cc;       // 2304

// ---------------------------------------------------------------------------
// helpers
// ---------------------------------------------------------------------------
__device__ inline v16h load2x8(const _Float16* __restrict__ p0,
                               const _Float16* __restrict__ p1) {
  v8h a = *reinterpret_cast<const v8h*>(p0);
  v8h b = *reinterpret_cast<const v8h*>(p1);
  v16h r;
#pragma unroll
  for (int j = 0; j < 8; ++j) { r[j] = a[j]; r[j + 8] = b[j]; }
  return r;
}

// One 16x16 f32 tile of A(row-major, lda) x Bt(row-major, ldb; Bt[n][k]),
// K multiple of 32.  A points at tile row 0, Bt at tile col-row 0.
__device__ inline v8f gemm_tile16(const _Float16* __restrict__ A, int lda,
                                  const _Float16* __restrict__ Bt, int ldb,
                                  int K, int lane) {
  const int r  = lane & 15;
  const int hh = lane >> 4;
  v8f acc = {};
  const _Float16* arow = A + r * lda;
  const _Float16* brow = Bt + r * ldb;
  for (int k0 = 0; k0 < K; k0 += 32) {
    // A frag: K = k0 + 16h + j, contiguous 16 halves
    v16h a = load2x8(arow + k0 + 16 * hh, arow + k0 + 16 * hh + 8);
    // B frag: K = k0 + 16h + j, contiguous 16 halves
    v16h b = load2x8(brow + k0 + 16 * hh, brow + k0 + 16 * hh + 8);
    acc = __builtin_amdgcn_wmma_f32_16x16x32_f16(false, a, false, b,
                                                 (short)0, acc, false, false);
  }
  return acc;
}

// ---------------------------------------------------------------------------
// prep kernels
// ---------------------------------------------------------------------------
__global__ void k_cvt_f16(const float* __restrict__ src,
                          _Float16* __restrict__ dst, int n) {
  int gid = blockIdx.x * blockDim.x + threadIdx.x;
  if (gid < n) dst[gid] = (_Float16)src[gid];
}

// wt[n][k] = (f16) w[k][n]   (w is [K x Nw] row major)
__global__ void k_transpose_cvt(const float* __restrict__ w,
                                _Float16* __restrict__ wt, int K, int Nw) {
  __shared__ float tile[16][17];
  int k0 = blockIdx.x * 16, n0 = blockIdx.y * 16;
  int tx = threadIdx.x & 15, ty = threadIdx.x >> 4;
  tile[ty][tx] = w[(k0 + ty) * Nw + (n0 + tx)];
  __syncthreads();
  wt[(n0 + ty) * K + (k0 + tx)] = (_Float16)tile[tx][ty];
}

// ---------------------------------------------------------------------------
// QKV projection: [8192 x 768] x [768 x 2304] + bias, scatter to
//   q  f16 [B,H,N,D]  (pre-scaled by D^-1/2)
//   k  f16 [B,H,N,D]
//   vT f16 [B,H,D,N]
// ---------------------------------------------------------------------------
__global__ void k_qkv_gemm(const _Float16* __restrict__ x16,
                           const _Float16* __restrict__ wt,
                           const float* __restrict__ bias,
                           _Float16* __restrict__ q,
                           _Float16* __restrict__ kmat,
                           _Float16* __restrict__ vT) {
  const int mt = blockIdx.x, nt = blockIdx.y;
  const int lane = threadIdx.x;
  v8f acc = gemm_tile16(x16 + (size_t)mt * 16 * Cc, Cc,
                        wt + (size_t)nt * 16 * Cc, Cc, Cc, lane);
  const int c   = nt * 16 + (lane & 15);
  const int s   = c / Cc;              // 0:q 1:k 2:v (uniform per block)
  const int rem = c % Cc;
  const int hh  = rem / Dd;
  const int dd  = rem % Dd;
  const float bv = bias[c];
#pragma unroll
  for (int g = 0; g < 8; ++g) {
    const int row = mt * 16 + g + 8 * (lane >> 4);   // 0..8191
    const int bat = row >> 11;                       // /2048
    const int nn  = row & (Nn - 1);
    const int bh  = bat * Hh + hh;
    const float val = acc[g] + bv;
    if (s == 0)      q[((size_t)bh * Nn + nn) * Dd + dd] = (_Float16)(val * 0.125f);
    else if (s == 1) kmat[((size_t)bh * Nn + nn) * Dd + dd] = (_Float16)val;
    else             vT[((size_t)bh * Dd + dd) * Nn + nn] = (_Float16)val;
  }
}

// ---------------------------------------------------------------------------
// Flash attention: one wave per (16 query rows, head, batch).
// Key tiles of 32.  Writes att f16 [B, N, C] (head-merged layout).
// ---------------------------------------------------------------------------
__global__ void k_flash_attn(const _Float16* __restrict__ q,
                             const _Float16* __restrict__ kmat,
                             const _Float16* __restrict__ vT,
                             const float* __restrict__ alibi,
                             const unsigned char* __restrict__ pmask,
                             _Float16* __restrict__ att) {
  const int qt   = blockIdx.x;          // 0..127
  const int bh   = blockIdx.y;          // 0..47
  const int bat  = bh / Hh;
  const int head = bh % Hh;
  const int lane = threadIdx.x;
  const int r    = lane & 15;
  const int hh   = lane >> 4;
  const int q0   = qt * 16;

  __shared__ _Float16 pl[16][32];       // P tile, C-layout -> A-layout bounce

  // Q fragments (row r of the tile), kept resident across all key tiles
  const _Float16* qrow = q + ((size_t)bh * Nn + q0 + r) * Dd;
  v16h aq0 = load2x8(qrow + 16 * hh,      qrow + 16 * hh + 8);
  v16h aq1 = load2x8(qrow + 32 + 16 * hh, qrow + 32 + 16 * hh + 8);

  v8f o0 = {}, o1 = {}, o2 = {}, o3 = {};
  float mrow[8], lrow[8];
#pragma unroll
  for (int g = 0; g < 8; ++g) { mrow[g] = -1e30f; lrow[g] = 0.0f; }

  for (int kt = 0; kt < Nn / 32; ++kt) {
    const int key0 = kt * 32;
    // ---- scores: two 16x16 tiles over 32 keys, K-dim = D = 64 (2 steps)
    v8f st[2];
#pragma unroll
    for (int t = 0; t < 2; ++t) {
      const _Float16* krow = kmat + ((size_t)bh * Nn + key0 + 16 * t + r) * Dd;
      v16h bk0 = load2x8(krow + 16 * hh,      krow + 16 * hh + 8);
      v16h bk1 = load2x8(krow + 32 + 16 * hh, krow + 32 + 16 * hh + 8);
      v8f z = {};
      z = __builtin_amdgcn_wmma_f32_16x16x32_f16(false, aq0, false, bk0,
                                                 (short)0, z, false, false);
      st[t] = __builtin_amdgcn_wmma_f32_16x16x32_f16(false, aq1, false, bk1,
                                                     (short)0, z, false, false);
    }
    // ---- alibi + padding mask + online softmax (C layout: row g+8h, col r)
    const unsigned char pm0 = pmask[bat * Nn + key0 + r];
    const unsigned char pm1 = pmask[bat * Nn + key0 + 16 + r];
#pragma unroll
    for (int g = 0; g < 8; ++g) {
      const long long arow = ((long long)bh * Nn + (q0 + g + 8 * hh)) * Nn + key0;
      float s0 = st[0][g] + alibi[arow + r];
      float s1 = st[1][g] + alibi[arow + 16 + r];
      if (pm0) s0 = -1e30f;
      if (pm1) s1 = -1e30f;
      float rmax = fmaxf(s0, s1);
#pragma unroll
      for (int off = 1; off < 16; off <<= 1)
        rmax = fmaxf(rmax, __shfl_xor(rmax, off, 32));
      const float mnew  = fmaxf(mrow[g], rmax);
      const float alpha = __expf(mrow[g] - mnew);
      const float p0 = __expf(s0 - mnew);
      const float p1 = __expf(s1 - mnew);
      float rs = p0 + p1;
#pragma unroll
      for (int off = 1; off < 16; off <<= 1)
        rs += __shfl_xor(rs, off, 32);
      lrow[g] = lrow[g] * alpha + rs;
      mrow[g] = mnew;
      o0[g] *= alpha; o1[g] *= alpha; o2[g] *= alpha; o3[g] *= alpha;
      pl[g + 8 * hh][r]      = (_Float16)p0;
      pl[g + 8 * hh][16 + r] = (_Float16)p1;
    }
    __syncthreads();
    // P tile in A layout (K over the 32 keys of this tile)
    v16h ap = load2x8(&pl[r][16 * hh], &pl[r][16 * hh + 8]);
    __syncthreads();
    // ---- P x V : one K=32 WMMA per 16-wide d tile; vT gives contiguous B frags
#pragma unroll
    for (int dt = 0; dt < 4; ++dt) {
      const _Float16* vrow =
          vT + ((size_t)bh * Dd + dt * 16 + r) * Nn + key0 + 16 * hh;
      v16h bv = load2x8(vrow, vrow + 8);
      v8f* op = (dt == 0) ? &o0 : (dt == 1) ? &o1 : (dt == 2) ? &o2 : &o3;
      *op = __builtin_amdgcn_wmma_f32_16x16x32_f16(false, ap, false, bv,
                                                   (short)0, *op, false, false);
    }
  }
  // ---- epilogue: divide by row sum, write att[b, n, head*64 + d] as f16
#pragma unroll
  for (int g = 0; g < 8; ++g) {
    const float inv = 1.0f / lrow[g];
    const int rown = q0 + g + 8 * hh;
    _Float16* dst = att + ((size_t)bat * Nn + rown) * Cc + head * Dd + r;
    dst[0]  = (_Float16)(o0[g] * inv);
    dst[16] = (_Float16)(o1[g] * inv);
    dst[32] = (_Float16)(o2[g] * inv);
    dst[48] = (_Float16)(o3[g] * inv);
  }
}

// ---------------------------------------------------------------------------
// Output projection: att[8192 x 768] x proj_w[768 x 768] + bias -> f32 out
// ---------------------------------------------------------------------------
__global__ void k_proj_gemm(const _Float16* __restrict__ att,
                            const _Float16* __restrict__ wt,
                            const float* __restrict__ bias,
                            float* __restrict__ out) {
  const int mt = blockIdx.x, nt = blockIdx.y;
  const int lane = threadIdx.x;
  v8f acc = gemm_tile16(att + (size_t)mt * 16 * Cc, Cc,
                        wt + (size_t)nt * 16 * Cc, Cc, Cc, lane);
  const int col = nt * 16 + (lane & 15);
  const float bv = bias[col];
#pragma unroll
  for (int g = 0; g < 8; ++g) {
    const int row = mt * 16 + g + 8 * (lane >> 4);
    out[(size_t)row * Cc + col] = acc[g] + bv;
  }
}

// ---------------------------------------------------------------------------
extern "C" void kernel_launch(void* const* d_in, const int* in_sizes, int n_in,
                              void* d_out, int out_size, void* d_ws, size_t ws_size,
                              hipStream_t stream) {
  const float*         x      = (const float*)d_in[0];
  const unsigned char* pmask  = (const unsigned char*)d_in[1];
  const float*         alibi  = (const float*)d_in[2];
  const float*         qkv_w  = (const float*)d_in[3];
  const float*         qkv_b  = (const float*)d_in[4];
  const float*         proj_w = (const float*)d_in[5];
  const float*         proj_b = (const float*)d_in[6];
  float*               out    = (float*)d_out;

  // workspace carve-up (all chunks are multiples of 256 B)
  const size_t nBNC = (size_t)Bb * Nn * Cc;       // 6,291,456
  char* ws = (char*)d_ws;
  _Float16* x16   = (_Float16*)ws;                ws += nBNC * 2;
  _Float16* q16   = (_Float16*)ws;                ws += nBNC * 2;
  _Float16* k16   = (_Float16*)ws;                ws += nBNC * 2;
  _Float16* vT16  = (_Float16*)ws;                ws += nBNC * 2;
  _Float16* att16 = (_Float16*)ws;                ws += nBNC * 2;
  _Float16* wqkvT = (_Float16*)ws;                ws += (size_t)C3 * Cc * 2;
  _Float16* wprjT = (_Float16*)ws;                ws += (size_t)Cc * Cc * 2;
  (void)ws_size; (void)in_sizes; (void)n_in; (void)out_size;

  // 1) convert x to f16
  k_cvt_f16<<<dim3((unsigned)((nBNC + 255) / 256)), dim3(256), 0, stream>>>(
      x, x16, (int)nBNC);
  // 2) transpose+convert weights
  k_transpose_cvt<<<dim3(Cc / 16, C3 / 16), dim3(256), 0, stream>>>(
      qkv_w, wqkvT, Cc, C3);
  k_transpose_cvt<<<dim3(Cc / 16, Cc / 16), dim3(256), 0, stream>>>(
      proj_w, wprjT, Cc, Cc);
  // 3) QKV projection (one wave per 16x16 tile)
  k_qkv_gemm<<<dim3(Bb * Nn / 16, C3 / 16), dim3(32), 0, stream>>>(
      x16, wqkvT, qkv_b, q16, k16, vT16);
  // 4) flash attention
  k_flash_attn<<<dim3(Nn / 16, Bb * Hh), dim3(32), 0, stream>>>(
      q16, k16, vT16, alibi, pmask, att16);
  // 5) output projection
  k_proj_gemm<<<dim3(Bb * Nn / 16, Cc / 16), dim3(32), 0, stream>>>(
      att16, wprjT, proj_b, out);
}
